// RewrittenSplitFF_32418413150501
// MI455X (gfx1250) — compile-verified, matrix operand order checked by reading, not run on hardware
//
#include <hip/hip_runtime.h>

typedef __bf16 bf16_t;
typedef __attribute__((ext_vector_type(16))) __bf16 v16bf;
typedef __attribute__((ext_vector_type(8)))  float  v8f;

#define DM 1024
#define NE 32
#define ES 128
#define TG 32

// Fragment helper: CDNA5 16-bit A/B fragment = 16 elements per lane; for lane
// group g = lane/16, element i maps to kk = 16*(i/8) + 8*g + (i%8) within a
// 32-wide K tile, i.e. two contiguous 8-element runs at kk = 8g and 16+8g.
__device__ __forceinline__ v16bf load_frag16(const bf16_t* base, int g) {
  const bf16_t* p0 = base + 8 * g;
  const bf16_t* p1 = base + 16 + 8 * g;
  v16bf f;
#pragma unroll
  for (int i = 0; i < 8; ++i) { f[i] = p0[i]; f[i + 8] = p1[i]; }
  return f;
}

// CDNA5 async LDS<->global DMA (ASYNCcnt-tracked). LDS operand is the 32-bit
// LDS byte offset (low 32 bits of a generic pointer to __shared__ memory).
__device__ __forceinline__ void async_load_b128(unsigned lds_off, const void* gaddr) {
  asm volatile("global_load_async_to_lds_b128 %0, %1, off"
               :: "v"(lds_off), "v"(gaddr) : "memory");
}
__device__ __forceinline__ void async_store_b128(void* gaddr, unsigned lds_off) {
  asm volatile("global_store_async_from_lds_b128 %0, %1, off"
               :: "v"(gaddr), "v"(lds_off) : "memory");
}
__device__ __forceinline__ void wait_async0() {
  asm volatile("s_wait_asynccnt 0x0" ::: "memory");
}

// Repack fp32 weights into bf16, B-fragment-native layout [ .. ][ktile][n][kk].
// mode 0: controller [1024,32]  -> cbc[kt(32)][n(32)][kk(32)]
// mode 1: f1 [1024,32,128]      -> f1c[e(32)][kt(32)][n(128)][kk(32)]
// mode 2: f2 [32,128,1024]      -> f2c[e(32)][kt(4)][n(1024)][kk(32)]
__global__ void prep_pack(const float* __restrict__ src, bf16_t* __restrict__ dst,
                          int mode, int n) {
  int F = blockIdx.x * blockDim.x + threadIdx.x;
  if (F >= n) return;
  int kk = F & 31;
  float v;
  if (mode == 0) {
    int nn = (F >> 5) & 31, kt = F >> 10;
    v = src[(kt * 32 + kk) * 32 + nn];
  } else if (mode == 1) {
    int nn = (F >> 5) & 127, kt = (F >> 12) & 31, e = F >> 17;
    v = src[(size_t)(kt * 32 + kk) * 4096 + e * 128 + nn];
  } else {
    int nn = (F >> 5) & 1023, kt = (F >> 15) & 3, e = F >> 17;
    v = src[(size_t)e * 131072 + (kt * 32 + kk) * 1024 + nn];
  }
  dst[F] = (bf16_t)v;
}

// Routing: per group (512 blocks, 4 waves). Converts the group's 32x1024 fp32
// tokens to bf16 in LDS, publishes the bf16 tile to global (async DMA from
// LDS, overlapped with WMMA), computes logits = tokens @ controller via bf16
// WMMA, then per-expert argmax + softmax prob of the winner.
__global__ __launch_bounds__(128) void routing_kernel(
    const float* __restrict__ x, const bf16_t* __restrict__ cbc,
    bf16_t* __restrict__ xbf, int* __restrict__ sel_tok,
    float* __restrict__ sel_p) {
  __shared__ bf16_t sX[32][DM];                // 64 KB, reused for logits after
  int G   = blockIdx.x;
  int tid = threadIdx.x;

  const float4* xg = (const float4*)(x + (size_t)G * (TG * DM));
  for (int i = tid; i < TG * DM / 4; i += 128) {
    float4 v = xg[i];
    int r = i >> 8, c = (i & 255) << 2;
    sX[r][c]     = (bf16_t)v.x; sX[r][c + 1] = (bf16_t)v.y;
    sX[r][c + 2] = (bf16_t)v.z; sX[r][c + 3] = (bf16_t)v.w;
  }
  __syncthreads();

  // Publish bf16 tile to xbf[G*32*DM ..] via async DMA straight out of LDS.
  {
    unsigned lds_base = (unsigned)(uintptr_t)&sX[0][0];
    bf16_t*  gdst     = xbf + (size_t)G * (TG * DM);
    for (int i = tid; i < TG * DM / 8; i += 128)          // 16B chunks
      async_store_b128(gdst + i * 8, lds_base + i * 16);
  }

  int w = tid >> 5, lane = tid & 31;
  int m0 = (w >> 1) * 16, n0 = (w & 1) * 16;
  v8f acc = {};
  for (int kt = 0; kt < 32; ++kt) {
    v16bf a = load_frag16(&sX[m0 + (lane & 15)][kt * 32], lane >> 4);
    v16bf b = load_frag16(cbc + (((kt * 32) + n0 + (lane & 15)) << 5), lane >> 4);
    acc = __builtin_amdgcn_wmma_f32_16x16x32_bf16(false, a, false, b,
                                                  (short)0, acc, false, false);
  }
  wait_async0();                                // DMA done reading LDS
  __syncthreads();                              // all waves done with sX
  float* sL = (float*)&sX[0][0];                // reuse LDS for 32x32 logits
#pragma unroll
  for (int v = 0; v < 8; ++v)
    sL[(m0 + v + 8 * (lane >> 4)) * 32 + (n0 + (lane & 15))] = acc[v];
  __syncthreads();

  if (tid < 32) {
    int e = tid;
    const float eps = 1e-6f / 31.0f;            // linspace(0,1e-6,32) step
    float best = -3.4e38f; int bt = 0;
    for (int t = 0; t < 32; ++t) {
      float l = sL[t * 32 + e] + (float)t * eps;
      if (l > best) { best = l; bt = t; }
    }
    float s = 0.f;
    for (int t = 0; t < 32; ++t)
      s += __expf(sL[t * 32 + e] + (float)t * eps - best);
    sel_tok[G * 32 + e] = bt;
    sel_p[G * 32 + e]   = 1.0f / s;             // softmax prob of argmax token
  }
}

// Per (expert, tile of 16 groups): async-DMA-gather the 16 selected bf16 rows
// into LDS, FF1 (WMMA bf16, K=1024) with the dispatch weight p folded into the
// epilogue (relu(p*(W x) + b) == relu(W (p x) + b) since p > 0), bias+ReLU,
// FF2 (WMMA bf16, K=128), scatter p*result atomically.
__global__ __launch_bounds__(128) void ffn_kernel(
    const bf16_t* __restrict__ xbf, const bf16_t* __restrict__ f1c,
    const bf16_t* __restrict__ f2c, const float* __restrict__ f1b,
    const int* __restrict__ sel_tok, const float* __restrict__ sel_p,
    float* __restrict__ out) {
  __shared__ bf16_t sA[16][DM];                 // 32 KB
  __shared__ bf16_t sI[16][ES];                 // 4 KB
  __shared__ int    sT[16];
  __shared__ float  sP[16];
  int mt = blockIdx.x, e = blockIdx.y;
  int tid = threadIdx.x;

  if (tid < 16) {
    int G = mt * 16 + tid;
    sT[tid] = sel_tok[G * 32 + e];
    sP[tid] = sel_p[G * 32 + e];
  }
  __syncthreads();

  // Gather 16 rows x 2KB of bf16 token data into LDS via async DMA.
  {
    unsigned lds_base = (unsigned)(uintptr_t)&sA[0][0];
    for (int i = tid; i < 16 * DM / 8; i += 128) {        // 16B chunks
      int r = i >> 7, chunk = i & 127;
      int G = mt * 16 + r;
      const bf16_t* gsrc = xbf + (size_t)(G * 32 + sT[r]) * DM + chunk * 8;
      async_load_b128(lds_base + i * 16, gsrc);
    }
  }
  wait_async0();
  __syncthreads();

  int w = tid >> 5, lane = tid & 31;

  // FF1: [16 x 1024] @ f1[:,e,:] -> [16 x 128]; each wave owns 32 columns.
  for (int sub = 0; sub < 2; ++sub) {
    int n0 = w * 32 + sub * 16;
    v8f acc = {};
    for (int kt = 0; kt < 32; ++kt) {
      v16bf a = load_frag16(&sA[lane & 15][kt * 32], lane >> 4);
      v16bf b = load_frag16(
          f1c + (((size_t)(e * 32 + kt) * 128 + n0 + (lane & 15)) << 5),
          lane >> 4);
      acc = __builtin_amdgcn_wmma_f32_16x16x32_bf16(false, a, false, b,
                                                    (short)0, acc, false, false);
    }
#pragma unroll
    for (int v = 0; v < 8; ++v) {
      int m = v + 8 * (lane >> 4);
      int n = n0 + (lane & 15);
      float val = acc[v] * sP[m] + f1b[e * 128 + n];      // dispatch weight p
      sI[m][n] = (bf16_t)fmaxf(val, 0.0f);
    }
  }
  __syncthreads();

  // FF2: [16 x 128] @ f2[e] -> [16 x 1024]; each wave owns 256 columns.
  for (int j = 0; j < 16; ++j) {
    int n0 = w * 256 + j * 16;
    v8f acc = {};
#pragma unroll
    for (int kt = 0; kt < 4; ++kt) {
      v16bf a = load_frag16(&sI[lane & 15][kt * 32], lane >> 4);
      v16bf b = load_frag16(
          f2c + (((size_t)(e * 4 + kt) * 1024 + n0 + (lane & 15)) << 5),
          lane >> 4);
      acc = __builtin_amdgcn_wmma_f32_16x16x32_bf16(false, a, false, b,
                                                    (short)0, acc, false, false);
    }
#pragma unroll
    for (int v = 0; v < 8; ++v) {
      int m   = v + 8 * (lane >> 4);
      int G   = mt * 16 + m;
      int col = n0 + (lane & 15);
      atomicAdd(out + (size_t)(G * 32 + sT[m]) * DM + col, sP[m] * acc[v]);
    }
  }
}

extern "C" void kernel_launch(void* const* d_in, const int* in_sizes, int n_in,
                              void* d_out, int out_size, void* d_ws, size_t ws_size,
                              hipStream_t stream) {
  const float* x          = (const float*)d_in[0];
  const float* controller = (const float*)d_in[1];
  const float* f1         = (const float*)d_in[2];
  const float* f2         = (const float*)d_in[3];
  const float* f1b        = (const float*)d_in[4];
  (void)in_sizes; (void)n_in; (void)ws_size;

  char* ws = (char*)d_ws;
  bf16_t* cbc     = (bf16_t*)ws;                                          // 64 KB
  bf16_t* f1c     = (bf16_t*)(ws + (1u << 16));                           // 8 MB
  bf16_t* f2c     = (bf16_t*)(ws + (1u << 16) + (8u << 20));              // 8 MB
  int*    sel_tok = (int*)  (ws + (1u << 16) + 2u * (8u << 20));          // 64 KB
  float*  sel_p   = (float*)(ws + 2u * (1u << 16) + 2u * (8u << 20));     // 64 KB
  bf16_t* xbf     = (bf16_t*)(ws + 3u * (1u << 16) + 2u * (8u << 20));    // 32 MB

  hipMemsetAsync(d_out, 0, (size_t)out_size * sizeof(float), stream);

  prep_pack<<<(32768   + 255) / 256, 256, 0, stream>>>(controller, cbc, 0, 32768);
  prep_pack<<<(4194304 + 255) / 256, 256, 0, stream>>>(f1, f1c, 1, 4194304);
  prep_pack<<<(4194304 + 255) / 256, 256, 0, stream>>>(f2, f2c, 2, 4194304);

  routing_kernel<<<512, 128, 0, stream>>>(x, cbc, xbf, sel_tok, sel_p);
  ffn_kernel<<<dim3(32, 32), 128, 0, stream>>>(xbf, f1c, f2c, f1b,
                                               sel_tok, sel_p, (float*)d_out);
}
